// BiLSTMTagger_46488726012485
// MI455X (gfx1250) — compile-verified, matrix operand order checked by reading, not run on hardware
//
#include <hip/hip_runtime.h>
#include <hip/hip_bf16.h>

// ---------------- types / helpers ----------------
typedef __attribute__((ext_vector_type(16))) _Float16 v16h;
typedef __attribute__((ext_vector_type(8)))  _Float16 v8h;
typedef __attribute__((ext_vector_type(8)))  float    v8f;

#define WMMA(acc, a, b)                                                        \
  (acc) = __builtin_amdgcn_wmma_f32_16x16x32_f16(false, (a), false, (b),       \
                                                 (short)0, (acc), false, false)

__device__ __forceinline__ float fsig(float x) { return 1.0f / (1.0f + __expf(-x)); }
__device__ __forceinline__ float ftanh(float x) { return 2.0f * fsig(2.0f * x) - 1.0f; }

// A/B fragment for v_wmma_f32_16x16x32_f16.
// Lane holds row (m0 + lane%16); half = lane/16 selects K groups:
//   VGPR0..3 : K = 8*half + 0..7      (16 contiguous bytes)
//   VGPR4..7 : K = 16 + 8*half + 0..7 (16 contiguous bytes)
__device__ __forceinline__ v16h load_frag16(const _Float16* __restrict__ rowptr,
                                            int k0, int half) {
  const _Float16* p = rowptr + k0 + (half << 3);
  v8h lo = *(const v8h*)p;
  v8h hi = *(const v8h*)(p + 16);
  return __builtin_shufflevector(lo, hi, 0, 1, 2, 3, 4, 5, 6, 7,
                                        8, 9, 10, 11, 12, 13, 14, 15);
}

// ---------------- small utility kernels ----------------
__global__ void k_f32_to_f16(const float* __restrict__ s, _Float16* __restrict__ d, int n) {
  for (int i = blockIdx.x * blockDim.x + threadIdx.x; i < n; i += gridDim.x * blockDim.x)
    d[i] = (_Float16)s[i];
}
__global__ void k_add_bias(const float* __restrict__ a, const float* __restrict__ b,
                           float* __restrict__ d, int n) {
  for (int i = blockIdx.x * blockDim.x + threadIdx.x; i < n; i += gridDim.x * blockDim.x)
    d[i] = a[i] + b[i];
}
__global__ void k_zero_h(_Float16* __restrict__ p, int n) {
  for (int i = blockIdx.x * blockDim.x + threadIdx.x; i < n; i += gridDim.x * blockDim.x)
    p[i] = (_Float16)0.0f;
}
__global__ void k_zero_f(float* __restrict__ p, int n) {
  for (int i = blockIdx.x * blockDim.x + threadIdx.x; i < n; i += gridDim.x * blockDim.x)
    p[i] = 0.0f;
}
// w_tag [50,1024] -> padded f16 [64,1024]; b_tag [50] -> padded f32 [64]
__global__ void k_tag_pad(const float* __restrict__ w_tag, const float* __restrict__ b_tag,
                          _Float16* __restrict__ wpad, float* __restrict__ bpad) {
  int i = blockIdx.x * blockDim.x + threadIdx.x; // 65536 threads
  int row = i >> 10, col = i & 1023;
  wpad[i] = (row < 50) ? (_Float16)w_tag[row * 1024 + col] : (_Float16)0.0f;
  if (i < 64) bpad[i] = (i < 50) ? b_tag[i] : 0.0f;
}
// word embedding gather -> cols [0,256) of ecat [16384,512]
__global__ void k_word_gather(const int* __restrict__ sent, const float* __restrict__ w_emb,
                              _Float16* __restrict__ ecat) {
  int i = blockIdx.x * blockDim.x + threadIdx.x; // 16384*256
  int row = i >> 8, col = i & 255;
  int idx = sent[row];
  ecat[(size_t)row * 512 + col] = (_Float16)w_emb[(size_t)idx * 256 + col];
}
// char embedding gather (padding_idx=0) -> ce [16][16384][64]
__global__ void k_char_gather(const int* __restrict__ csent, const float* __restrict__ w_char,
                              _Float16* __restrict__ ce) {
  int i = blockIdx.x * blockDim.x + threadIdx.x; // 16*16384*64 = 2^24
  int cd = i & 63;
  int bs = (i >> 6) & 16383;
  int w  = i >> 20;
  int idx = csent[bs * 16 + w];
  float v = (idx == 0) ? 0.0f : w_char[idx * 64 + cd];
  ce[i] = (_Float16)v;
}

// ---------------- fused bidirectional LSTM step ----------------
// blockIdx.z selects direction (fwd/bwd are independent -> run concurrently).
// One wave computes an (MT*16) x 16 tile of all 4 gates:
// gates = x@wih.T + hprev@whh.T + bias ; c = sig(f)*c + sig(i)*tanh(g) ; h = sig(o)*tanh(c)
// B (weight) fragments are shared across the MT row tiles; all fragments of a
// K-chunk are loaded before any WMMA issues so VMEM latency overlaps matrix ops.
struct StepArgs {
  const _Float16* x[2];
  const _Float16* hprev[2];
  float*          cbuf[2];     // [N,Hh] f32, updated in place
  const _Float16* wih[2];      // [4*Hh, Din]
  const _Float16* whh[2];      // [4*Hh, Hh]
  const float*    bias[2];     // [4*Hh] (bih+bhh)
  _Float16*       hout[2];
  _Float16*       seq[2];      // optional extra copy of h (may be null)
  int ldx, Din;
  int ldh[2];
  int ldhout, Hh, seq_ld;
};

template <int MT>
__global__ void __launch_bounds__(32) k_lstm_step(StepArgs P) {
  const int dir = blockIdx.z;
  const int lane = threadIdx.x;
  const int l16 = lane & 15, half = lane >> 4;
  const int m0 = blockIdx.x * (16 * MT);
  const int n0 = blockIdx.y << 4;
  const int Hh = P.Hh;

  v8f acc[MT][4];
#pragma unroll
  for (int i = 0; i < MT; ++i)
#pragma unroll
    for (int g = 0; g < 4; ++g) acc[i][g] = v8f{};

  // --- x @ wih.T ---
  {
    const _Float16* arow[MT];
#pragma unroll
    for (int i = 0; i < MT; ++i) arow[i] = P.x[dir] + (size_t)(m0 + 16 * i + l16) * P.ldx;
    const _Float16* wrow = P.wih[dir] + (size_t)(n0 + l16) * P.Din;
    const size_t gstride = (size_t)Hh * P.Din;
    for (int k0 = 0; k0 < P.Din; k0 += 32) {
      v16h a[MT], b[4];
#pragma unroll
      for (int i = 0; i < MT; ++i) a[i] = load_frag16(arow[i], k0, half);
#pragma unroll
      for (int g = 0; g < 4; ++g) b[g] = load_frag16(wrow + (size_t)g * gstride, k0, half);
#pragma unroll
      for (int g = 0; g < 4; ++g)
#pragma unroll
        for (int i = 0; i < MT; ++i) WMMA(acc[i][g], a[i], b[g]);
    }
  }
  // --- hprev @ whh.T ---
  {
    const _Float16* arow[MT];
#pragma unroll
    for (int i = 0; i < MT; ++i) arow[i] = P.hprev[dir] + (size_t)(m0 + 16 * i + l16) * P.ldh[dir];
    const _Float16* wrow = P.whh[dir] + (size_t)(n0 + l16) * Hh;
    const size_t gstride = (size_t)Hh * Hh;
    for (int k0 = 0; k0 < Hh; k0 += 32) {
      v16h a[MT], b[4];
#pragma unroll
      for (int i = 0; i < MT; ++i) a[i] = load_frag16(arow[i], k0, half);
#pragma unroll
      for (int g = 0; g < 4; ++g) b[g] = load_frag16(wrow + (size_t)g * gstride, k0, half);
#pragma unroll
      for (int g = 0; g < 4; ++g)
#pragma unroll
        for (int i = 0; i < MT; ++i) WMMA(acc[i][g], a[i], b[g]);
    }
  }
  // --- pointwise cell update ---
  const int col = n0 + l16;
  const float* bias = P.bias[dir];
  float* cbuf = P.cbuf[dir];
  _Float16* hout = P.hout[dir];
  _Float16* seq = P.seq[dir];
  const float bi = bias[col];
  const float bf = bias[Hh + col];
  const float bg = bias[2 * Hh + col];
  const float bo = bias[3 * Hh + col];
#pragma unroll
  for (int i = 0; i < MT; ++i) {
#pragma unroll
    for (int r = 0; r < 8; ++r) {
      int row = m0 + 16 * i + r + (half << 3);
      size_t cidx = (size_t)row * Hh + col;
      float iv = fsig(acc[i][0][r] + bi);
      float fv = fsig(acc[i][1][r] + bf);
      float gv = ftanh(acc[i][2][r] + bg);
      float ov = fsig(acc[i][3][r] + bo);
      float cv = fv * cbuf[cidx] + iv * gv;
      cbuf[cidx] = cv;
      float hv = ov * ftanh(cv);
      hout[(size_t)row * P.ldhout + col] = (_Float16)hv;
      if (seq) seq[(size_t)row * P.seq_ld + col] = (_Float16)hv;
    }
  }
}

// ---------------- final projection: ecat[16384,512] @ w_final.T -> xs [S][B][256] ----------------
__global__ void __launch_bounds__(32)
k_gemm_final(const _Float16* __restrict__ A, const _Float16* __restrict__ W,
             const float* __restrict__ bias, _Float16* __restrict__ out) {
  const int lane = threadIdx.x;
  const int l16 = lane & 15, half = lane >> 4;
  const int m0 = blockIdx.x << 6, n0 = blockIdx.y << 4;
  const _Float16* arow[4];
#pragma unroll
  for (int i = 0; i < 4; ++i) arow[i] = A + (size_t)(m0 + 16 * i + l16) * 512;
  const _Float16* wrow = W + (size_t)(n0 + l16) * 512;
  v8f acc[4] = {v8f{}, v8f{}, v8f{}, v8f{}};
#pragma unroll 2
  for (int k0 = 0; k0 < 512; k0 += 32) {
    v16h a[4];
#pragma unroll
    for (int i = 0; i < 4; ++i) a[i] = load_frag16(arow[i], k0, half);
    v16h w = load_frag16(wrow, k0, half);
#pragma unroll
    for (int i = 0; i < 4; ++i) WMMA(acc[i], a[i], w);
  }
  const int col = n0 + l16;
  const float bv = bias[col];
#pragma unroll
  for (int i = 0; i < 4; ++i) {
#pragma unroll
    for (int r = 0; r < 8; ++r) {
      int row = m0 + 16 * i + r + (half << 3); // row = b*256 + s
      int bb = row >> 8, ss = row & 255;
      out[(size_t)(ss * 64 + bb) * 256 + col] = (_Float16)(acc[i][r] + bv);
    }
  }
}

// ---------------- tag projection: o2 [16384,1024] @ w_tag.T -> d_out [B][S][50] f32 ----------------
__global__ void __launch_bounds__(32)
k_gemm_tag(const _Float16* __restrict__ A, const _Float16* __restrict__ W,
           const float* __restrict__ bias, float* __restrict__ out) {
  const int lane = threadIdx.x;
  const int l16 = lane & 15, half = lane >> 4;
  const int m0 = blockIdx.x << 6, n0 = blockIdx.y << 4;
  const _Float16* arow[4];
#pragma unroll
  for (int i = 0; i < 4; ++i) arow[i] = A + (size_t)(m0 + 16 * i + l16) * 1024;
  const _Float16* wrow = W + (size_t)(n0 + l16) * 1024;
  v8f acc[4] = {v8f{}, v8f{}, v8f{}, v8f{}};
#pragma unroll 2
  for (int k0 = 0; k0 < 1024; k0 += 32) {
    v16h a[4];
#pragma unroll
    for (int i = 0; i < 4; ++i) a[i] = load_frag16(arow[i], k0, half);
    v16h w = load_frag16(wrow, k0, half);
#pragma unroll
    for (int i = 0; i < 4; ++i) WMMA(acc[i], a[i], w);
  }
  const int col = n0 + l16;
  if (col < 50) {
    const float bv = bias[col];
#pragma unroll
    for (int i = 0; i < 4; ++i) {
#pragma unroll
      for (int r = 0; r < 8; ++r) {
        int row = m0 + 16 * i + r + (half << 3); // row = s*64 + b
        int bb = row & 63, ss = row >> 6;
        out[((size_t)bb * 256 + ss) * 50 + col] = acc[i][r] + bv;
      }
    }
  }
}

// ---------------- host ----------------
static inline size_t align256(size_t x) { return (x + 255) & ~(size_t)255; }
static inline int nblk(int n) { int b = (n + 255) / 256; return b; }

extern "C" void kernel_launch(void* const* d_in, const int* in_sizes, int n_in,
                              void* d_out, int out_size, void* d_ws, size_t ws_size,
                              hipStream_t stream) {
  (void)in_sizes; (void)n_in; (void)out_size; (void)ws_size;
  const int* sentence = (const int*)d_in[0];
  const int* csent    = (const int*)d_in[1];
  const float* w_emb  = (const float*)d_in[2];
  const float* w_char = (const float*)d_in[3];
  const float* w_final = (const float*)d_in[28];
  const float* b_final = (const float*)d_in[29];
  const float* w_tag   = (const float*)d_in[30];
  const float* b_tag   = (const float*)d_in[31];

  struct CellDim { int din, h; };
  const CellDim cd6[6] = {{64,128},{64,128},{256,512},{256,512},{1024,512},{1024,512}};

  // ---- workspace layout ----
  char* ws = (char*)d_ws;
  size_t off = 0;
  auto alloc = [&](size_t bytes) -> void* { void* p = ws + off; off = align256(off + bytes); return p; };

  _Float16* wih_h[6]; _Float16* whh_h[6]; float* bias_f[6];
  for (int j = 0; j < 6; ++j) {
    wih_h[j]  = (_Float16*)alloc((size_t)4 * cd6[j].h * cd6[j].din * 2);
    whh_h[j]  = (_Float16*)alloc((size_t)4 * cd6[j].h * cd6[j].h * 2);
    bias_f[j] = (float*)alloc((size_t)4 * cd6[j].h * 4);
  }
  _Float16* wfin_h = (_Float16*)alloc((size_t)256 * 512 * 2);
  _Float16* wtag_h = (_Float16*)alloc((size_t)64 * 1024 * 2);
  float*    btag_f = (float*)alloc(64 * 4);
  _Float16* ce   = (_Float16*)alloc((size_t)16 * 16384 * 64 * 2);   // [W][BS][64]
  _Float16* ecat = (_Float16*)alloc((size_t)16384 * 512 * 2);       // [BS][512]
  _Float16* xs   = (_Float16*)alloc((size_t)16384 * 256 * 2);       // [S][B][256]
  _Float16* o1   = (_Float16*)alloc((size_t)256 * 64 * 1024 * 2);   // [S][B][1024]
  _Float16* o2   = (_Float16*)alloc((size_t)256 * 64 * 1024 * 2);
  _Float16* ch_h0[2]; _Float16* ch_h1[2]; float* ch_c[2];
  for (int d = 0; d < 2; ++d) {
    ch_h0[d] = (_Float16*)alloc((size_t)16384 * 128 * 2);
    ch_h1[d] = (_Float16*)alloc((size_t)16384 * 128 * 2);
    ch_c[d]  = (float*)alloc((size_t)16384 * 128 * 4);
  }
  _Float16* wz = (_Float16*)alloc((size_t)64 * 512 * 2);
  float* wc[2];
  wc[0] = (float*)alloc((size_t)64 * 512 * 4);
  wc[1] = (float*)alloc((size_t)64 * 512 * 4);

  // ---- weight conversion / bias fusion ----
  for (int j = 0; j < 6; ++j) {
    int nwi = 4 * cd6[j].h * cd6[j].din, nwh = 4 * cd6[j].h * cd6[j].h, nb = 4 * cd6[j].h;
    k_f32_to_f16<<<nblk(nwi), 256, 0, stream>>>((const float*)d_in[4 + 4*j], wih_h[j], nwi);
    k_f32_to_f16<<<nblk(nwh), 256, 0, stream>>>((const float*)d_in[5 + 4*j], whh_h[j], nwh);
    k_add_bias  <<<nblk(nb),  256, 0, stream>>>((const float*)d_in[6 + 4*j],
                                                (const float*)d_in[7 + 4*j], bias_f[j], nb);
  }
  k_f32_to_f16<<<nblk(256 * 512), 256, 0, stream>>>(w_final, wfin_h, 256 * 512);
  k_tag_pad<<<256, 256, 0, stream>>>(w_tag, b_tag, wtag_h, btag_f);

  // ---- embedding gathers ----
  k_word_gather<<<16384, 256, 0, stream>>>(sentence, w_emb, ecat);
  k_char_gather<<<65536, 256, 0, stream>>>(csent, w_char, ce);

  // ---- char-level BiLSTM (batch 16384, Hh=128, 16 steps, both directions fused, MT=4) ----
  for (int d = 0; d < 2; ++d) {
    k_zero_h<<<4096, 256, 0, stream>>>(ch_h0[d], 16384 * 128);
    k_zero_f<<<4096, 256, 0, stream>>>(ch_c[d], 16384 * 128);
  }
  for (int tt = 0; tt < 16; ++tt) {
    StepArgs P;
    P.ldx = 64; P.Din = 64; P.ldhout = 128; P.Hh = 128; P.seq_ld = 512;
    int t[2] = {tt, 15 - tt};       // dir0 forward, dir1 backward
    for (int d = 0; d < 2; ++d) {
      P.x[d]     = ce + (size_t)t[d] * 16384 * 64;
      P.hprev[d] = (tt & 1) ? ch_h1[d] : ch_h0[d];
      P.hout[d]  = (tt & 1) ? ch_h0[d] : ch_h1[d];
      P.ldh[d]   = 128;
      P.cbuf[d]  = ch_c[d];
      P.wih[d]   = wih_h[d]; P.whh[d] = whh_h[d]; P.bias[d] = bias_f[d];
      P.seq[d]   = (tt == 15) ? (ecat + 256 + 128 * d) : nullptr; // hf/hb -> ecat cols
    }
    k_lstm_step<4><<<dim3(256, 8, 2), 32, 0, stream>>>(P);
  }

  // ---- final embedding projection -> xs [S][B][256] ----
  k_gemm_final<<<dim3(256, 16), 32, 0, stream>>>(ecat, wfin_h, b_final, xs);

  // ---- word-level BiLSTM layers (batch 64, Hh=512, 256 steps, directions fused, MT=2) ----
  k_zero_h<<<128, 256, 0, stream>>>(wz, 64 * 512);

  // layer 1: xs -> o1 (fwd cols [0,512), bwd cols [512,1024))
  k_zero_f<<<128, 256, 0, stream>>>(wc[0], 64 * 512);
  k_zero_f<<<128, 256, 0, stream>>>(wc[1], 64 * 512);
  for (int tt = 0; tt < 256; ++tt) {
    StepArgs P;
    P.ldx = 256; P.Din = 256; P.ldhout = 1024; P.Hh = 512; P.seq_ld = 0;
    int t[2] = {tt, 255 - tt};
    for (int d = 0; d < 2; ++d) {
      P.x[d]     = xs + (size_t)t[d] * 16384;
      P.hprev[d] = tt ? (o1 + (size_t)(t[d] + (d ? 1 : -1)) * 65536 + 512 * d) : wz;
      P.ldh[d]   = tt ? 1024 : 512;
      P.hout[d]  = o1 + (size_t)t[d] * 65536 + 512 * d;
      P.cbuf[d]  = wc[d];
      P.wih[d]   = wih_h[2 + d]; P.whh[d] = whh_h[2 + d]; P.bias[d] = bias_f[2 + d];
      P.seq[d]   = nullptr;
    }
    k_lstm_step<2><<<dim3(2, 32, 2), 32, 0, stream>>>(P);
  }
  // layer 2: o1 -> o2
  k_zero_f<<<128, 256, 0, stream>>>(wc[0], 64 * 512);
  k_zero_f<<<128, 256, 0, stream>>>(wc[1], 64 * 512);
  for (int tt = 0; tt < 256; ++tt) {
    StepArgs P;
    P.ldx = 1024; P.Din = 1024; P.ldhout = 1024; P.Hh = 512; P.seq_ld = 0;
    int t[2] = {tt, 255 - tt};
    for (int d = 0; d < 2; ++d) {
      P.x[d]     = o1 + (size_t)t[d] * 65536;
      P.hprev[d] = tt ? (o2 + (size_t)(t[d] + (d ? 1 : -1)) * 65536 + 512 * d) : wz;
      P.ldh[d]   = tt ? 1024 : 512;
      P.hout[d]  = o2 + (size_t)t[d] * 65536 + 512 * d;
      P.cbuf[d]  = wc[d];
      P.wih[d]   = wih_h[4 + d]; P.whh[d] = whh_h[4 + d]; P.bias[d] = bias_f[4 + d];
      P.seq[d]   = nullptr;
    }
    k_lstm_step<2><<<dim3(2, 32, 2), 32, 0, stream>>>(P);
  }

  // ---- tag projection -> d_out [B][S][50] f32 ----
  k_gemm_tag<<<dim3(256, 4), 32, 0, stream>>>(o2, wtag_h, btag_f, (float*)d_out);
}
